// REC_STAGE_18605798326274
// MI455X (gfx1250) — compile-verified, matrix operand order checked by reading, not run on hardware
//
#include <hip/hip_runtime.h>
#include <stdint.h>
#include <stddef.h>

// ---------------------------------------------------------------------------
// MI455X (gfx1250): bf16 WMMA GEMM/conv cores, fused epilogues, optional
// CDNA5 async global->LDS staging (guarded by __has_builtin so the file
// compiles on toolchains without the async builtins).
// ---------------------------------------------------------------------------

typedef __attribute__((ext_vector_type(16))) __bf16 v16bf;
typedef __attribute__((ext_vector_type(8)))  float  v8f;

#if defined(__has_builtin)
#if __has_builtin(__builtin_amdgcn_global_load_async_to_lds_b128)
#define HAVE_ASYNC_LDS 1
#endif
#endif

#ifdef HAVE_ASYNC_LDS
// Builtin signature (from toolchain diagnostic): param0 is
// 'int __attribute__((vector_size(16))) __device__ *' i.e. v4i in AS1.
typedef int v4i __attribute__((vector_size(16)));
typedef __attribute__((address_space(1))) const void* gas_cvp;
typedef __attribute__((address_space(1))) v4i*        gas_v4p;
typedef __attribute__((address_space(3))) void*       las_vp;
typedef __attribute__((address_space(3))) v4i*        las_v4p;
__device__ __forceinline__ void async_cp16(const void* g, void* l) {
  __builtin_amdgcn_global_load_async_to_lds_b128(
      (gas_v4p)(gas_cvp)g, (las_v4p)(las_vp)l, 0, 0);
}
__device__ __forceinline__ void async_wait0() {
#if __has_builtin(__builtin_amdgcn_s_wait_asynccnt)
  __builtin_amdgcn_s_wait_asynccnt(0);
#else
  asm volatile("s_wait_asynccnt 0x0" ::: "memory");
#endif
}
#endif

union FragU { v16bf v; uint32_t u[8]; unsigned short s[16]; };

__device__ __forceinline__ unsigned short f2bf(float x) {
  union { float f; uint32_t u; } c; c.f = x;
  uint32_t u = c.u + 0x7fffu + ((c.u >> 16) & 1u);   // round-to-nearest-even
  return (unsigned short)(u >> 16);
}
__device__ __forceinline__ float act_apply(float v, int act) {
  if (act == 1) v = fmaxf(v, 0.f);
  else if (act == 2) v = (v > 0.f) ? v : (__expf(v) - 1.f);   // ELU
  else if (act == 3) v = 1.f / (1.f + __expf(-v));            // sigmoid
  return v;
}

// 16(row)x32(K) bf16 fragment from LDS tile, row pitch 40 ushorts.
// CDNA5 ISA 7.12.2 16-bit A layout: half = lane>>4 -> K+8, pair p -> K=2p /
// 16+2(p-4), two elements per VGPR.
__device__ __forceinline__ v16bf load_frag(const unsigned short* tile) {
  int lane = threadIdx.x & 31;
  int m = lane & 15, half = lane >> 4;
  const unsigned short* row = tile + m * 40;
  FragU f;
#pragma unroll
  for (int p = 0; p < 8; ++p) {
    int k0 = ((p < 4) ? (2 * p) : (16 + 2 * (p - 4))) + (half ? 8 : 0);
    f.u[p] = *(const uint32_t*)(row + k0);
  }
  return f.v;
}

// Copy one 16-ushort chunk (32B) global->LDS (async when available).
__device__ __forceinline__ void stage32(const unsigned short* g, unsigned short* l) {
#ifdef HAVE_ASYNC_LDS
  async_cp16(g, l);
  async_cp16(g + 8, l + 8);
#else
  const uint4* s = (const uint4*)g;
  uint4* d = (uint4*)l;
  d[0] = s[0]; d[1] = s[1];
#endif
}
__device__ __forceinline__ void zero32(unsigned short* l) {
  uint4 z = make_uint4(0u, 0u, 0u, 0u);
  uint4* d = (uint4*)l;
  d[0] = z; d[1] = z;
}

// ---------------------------------------------------------------------------
// WMMA GEMM: C[M,N] = act(A[M,K](bf16) * W + bias)
//   b_kn==0 : W (N,K) row-major   b_kn==1 : W (K,N) row-major
// Block tile 128x128, 8 waves, each wave 32(M)x64(N) -> 8 WMMA / K-step.
// ---------------------------------------------------------------------------
__global__ void wmma_gemm(const unsigned short* __restrict__ A,
                          const unsigned short* __restrict__ B,
                          const float* __restrict__ bias,
                          float* __restrict__ Cf,
                          unsigned short* __restrict__ Cbf,
                          int M, int N, int K, int b_kn, int act,
                          long long sA, long long sB, long long sC) {
  __shared__ __align__(16) unsigned short As[128 * 40];
  __shared__ __align__(16) unsigned short Bs[128 * 40];
  const int tid = threadIdx.x;
  const int mBase = blockIdx.y * 128;
  const int nBase = blockIdx.x * 128;
  A += (size_t)blockIdx.z * sA;
  B += (size_t)blockIdx.z * sB;
  const size_t cOff = (size_t)blockIdx.z * sC;
  v8f acc[2][4] = {};
  const int sr = tid >> 1, scg = (tid & 1) << 4;   // staging row / col-chunk
  for (int k0 = 0; k0 < K; k0 += 32) {
    if (k0 + 32 < K && tid < 128 && mBase + tid < M)
      __builtin_prefetch(&A[(size_t)(mBase + tid) * K + k0 + 32], 0, 3);
    const bool vecok = (k0 + 32 <= K);
    if (vecok) {
      if (mBase + sr < M) stage32(A + (size_t)(mBase + sr) * K + k0 + scg,
                                  As + sr * 40 + scg);
      else                zero32(As + sr * 40 + scg);
    } else {
      for (int i = tid; i < 128 * 32; i += 256) {
        int r = i >> 5, c = i & 31;
        unsigned short v = 0;
        if (mBase + r < M && k0 + c < K) v = A[(size_t)(mBase + r) * K + k0 + c];
        As[r * 40 + c] = v;
      }
    }
    if (b_kn == 0 && vecok) {
      if (nBase + sr < N) stage32(B + (size_t)(nBase + sr) * K + k0 + scg,
                                  Bs + sr * 40 + scg);
      else                zero32(Bs + sr * 40 + scg);
    } else {
      for (int i = tid; i < 128 * 32; i += 256) {
        int r = i >> 5, c = i & 31;
        unsigned short v = 0;
        if (nBase + r < N && k0 + c < K)
          v = b_kn ? B[(size_t)(k0 + c) * N + nBase + r]
                   : B[(size_t)(nBase + r) * K + k0 + c];
        Bs[r * 40 + c] = v;
      }
    }
#ifdef HAVE_ASYNC_LDS
    async_wait0();
#endif
    __syncthreads();
    {
      int wave = tid >> 5, wm = wave & 3, wn = wave >> 2;
      v16bf a0 = load_frag(As + (wm * 32) * 40);
      v16bf a1 = load_frag(As + (wm * 32 + 16) * 40);
#pragma unroll
      for (int nn = 0; nn < 4; ++nn) {
        v16bf b = load_frag(Bs + (wn * 64 + nn * 16) * 40);
        acc[0][nn] = __builtin_amdgcn_wmma_f32_16x16x32_bf16(
            false, a0, false, b, (short)0, acc[0][nn], false, false);
        acc[1][nn] = __builtin_amdgcn_wmma_f32_16x16x32_bf16(
            false, a1, false, b, (short)0, acc[1][nn], false, false);
      }
    }
    __syncthreads();
  }
  int lane = tid & 31, wave = tid >> 5, wm = wave & 3, wn = wave >> 2;
  int half = lane >> 4, nl = lane & 15;
  for (int mm = 0; mm < 2; ++mm) {
    for (int nn = 0; nn < 4; ++nn) {
      int n = nBase + wn * 64 + nn * 16 + nl;
      if (n >= N) continue;
#pragma unroll
      for (int r = 0; r < 8; ++r) {
        int m = mBase + wm * 32 + mm * 16 + r + half * 8;
        if (m >= M) continue;
        float v = acc[mm][nn][r];
        if (bias) v += bias[n];
        v = act_apply(v, act);
        size_t ci = cOff + (size_t)m * N + n;
        if (Cf) Cf[ci] = v;
        if (Cbf) Cbf[ci] = f2bf(v);
      }
    }
  }
}

// ---------------------------------------------------------------------------
// 3x3 conv (pad 1) as implicit GEMM. Weights pre-packed bf16 (kyx, O, C) so
// the weight stage is a contiguous row copy (async-eligible). ups=1 fuses a
// 2x nearest upsample on the input coordinates. Epilogue: +bias -> BN -> act
// -> +resid, f32 and/or bf16 out.
// ---------------------------------------------------------------------------
__global__ void wmma_conv3x3(const unsigned short* __restrict__ X,   // bf16 NCHW
                             const unsigned short* __restrict__ Wbf, // (9,O,C) bf16
                             const float* __restrict__ cb,
                             const float* __restrict__ bng,
                             const float* __restrict__ bnb,
                             const float* __restrict__ resid,
                             float* __restrict__ Yf,
                             unsigned short* __restrict__ Ybf,
                             int C, int O, int Hin, int Win, int Ho, int Wo,
                             int stride, int ups, int act) {
  __shared__ __align__(16) unsigned short Ws[128 * 40];
  __shared__ __align__(16) unsigned short Xs[128 * 40];
  const int tid = threadIdx.x;
  const int img = blockIdx.z;
  const int oBase = blockIdx.y * 128;
  const int pBase = blockIdx.x * 128;
  const int P = Ho * Wo;
  const int Hf = ups ? 2 * Hin : Hin;
  const int Wf = ups ? 2 * Win : Win;
  const int sr = tid >> 1, scg = (tid & 1) << 4;
  v8f acc[2][4] = {};
  for (int kyx = 0; kyx < 9; ++kyx) {
    int ky = kyx / 3, kx = kyx % 3;
    for (int c0 = 0; c0 < C; c0 += 32) {
      if (oBase + sr < O)
        stage32(Wbf + ((size_t)kyx * O + oBase + sr) * C + c0 + scg,
                Ws + sr * 40 + scg);
      else
        zero32(Ws + sr * 40 + scg);
      for (int i = tid; i < 128 * 32; i += 256) {
        int p = i >> 5, c = i & 31;
        unsigned short v = 0;
        int pp = pBase + p;
        if (pp < P && c0 + c < C) {
          int y = pp / Wo, x = pp % Wo;
          int iy = y * stride - 1 + ky;
          int ix = x * stride - 1 + kx;
          if (iy >= 0 && iy < Hf && ix >= 0 && ix < Wf) {
            if (ups) { iy >>= 1; ix >>= 1; }
            v = X[(((size_t)img * C + (c0 + c)) * Hin + iy) * Win + ix];
          }
        }
        Xs[p * 40 + c] = v;
      }
#ifdef HAVE_ASYNC_LDS
      async_wait0();
#endif
      __syncthreads();
      {
        int wave = tid >> 5, wm = wave & 3, wn = wave >> 2;
        v16bf a0 = load_frag(Ws + (wm * 32) * 40);
        v16bf a1 = load_frag(Ws + (wm * 32 + 16) * 40);
#pragma unroll
        for (int nn = 0; nn < 4; ++nn) {
          v16bf b = load_frag(Xs + (wn * 64 + nn * 16) * 40);
          acc[0][nn] = __builtin_amdgcn_wmma_f32_16x16x32_bf16(
              false, a0, false, b, (short)0, acc[0][nn], false, false);
          acc[1][nn] = __builtin_amdgcn_wmma_f32_16x16x32_bf16(
              false, a1, false, b, (short)0, acc[1][nn], false, false);
        }
      }
      __syncthreads();
    }
  }
  int lane = tid & 31, wave = tid >> 5, wm = wave & 3, wn = wave >> 2;
  int half = lane >> 4, nl = lane & 15;
  for (int mm = 0; mm < 2; ++mm) {
    for (int nn = 0; nn < 4; ++nn) {
      int p = pBase + wn * 64 + nn * 16 + nl;
      if (p >= P) continue;
#pragma unroll
      for (int r = 0; r < 8; ++r) {
        int o = oBase + wm * 32 + mm * 16 + r + half * 8;
        if (o >= O) continue;
        float v = acc[mm][nn][r] + cb[o];
        if (bng) v = v * bng[o] + bnb[o];
        v = act_apply(v, act);
        size_t oi = ((size_t)img * O + o) * P + p;
        if (resid) v += resid[oi];
        if (Yf) Yf[oi] = v;
        if (Ybf) Ybf[oi] = f2bf(v);
      }
    }
  }
}

// ------------------------- small VALU kernels ------------------------------

__global__ void pack_bf16_k(const float* __restrict__ X, unsigned short* __restrict__ Y,
                            long long n) {
  long long i = (long long)blockIdx.x * blockDim.x + threadIdx.x;
  if (i < n) Y[i] = f2bf(X[i]);
}

// f32 (O,C,3,3) -> bf16 (kyx,O,C)
__global__ void repack_convw_k(const float* __restrict__ W, unsigned short* __restrict__ Y,
                               int O, int C) {
  long long i = (long long)blockIdx.x * blockDim.x + threadIdx.x;
  long long tot = 9LL * O * C;
  if (i >= tot) return;
  int c = (int)(i % C); long long t = i / C;
  int o = (int)(t % O); int kyx = (int)(t / O);
  Y[i] = f2bf(W[((size_t)o * C + c) * 9 + kyx]);
}

__global__ void gate_k(const float* __restrict__ a, const float* __restrict__ b,
                       unsigned short* __restrict__ outb, long long n) {
  long long i = (long long)blockIdx.x * blockDim.x + threadIdx.x;
  if (i < n) outb[i] = f2bf(a[i] * (1.f / (1.f + __expf(-b[i]))));
}

// (n,D,P) -> (n,P,D)
__global__ void t_ndp_npd(const float* __restrict__ X, float* __restrict__ Yf,
                          unsigned short* __restrict__ Ybf, int NN, int D, int P) {
  long long i = (long long)blockIdx.x * blockDim.x + threadIdx.x;
  long long tot = (long long)NN * D * P;
  if (i >= tot) return;
  int d = (int)(i % D); long long t = i / D;
  int p = (int)(t % P); int n = (int)(t / P);
  float v = X[((size_t)n * D + d) * P + p];
  if (Yf)  Yf[i] = v;
  if (Ybf) Ybf[i] = f2bf(v);
}

// (n,S,D) tokens -> (S,n,D) tokens
__global__ void t_ns_sn(const float* __restrict__ X, float* __restrict__ Y,
                        int NN, int S, int D) {
  long long i = (long long)blockIdx.x * blockDim.x + threadIdx.x;
  long long tot = (long long)NN * S * D;
  if (i >= tot) return;
  int d = (int)(i % D); long long t = i / D;
  int n = (int)(t % NN); int s = (int)(t / NN);
  Y[i] = X[((size_t)n * S + s) * D + d];
}

// (S,n,D) -> (n,D,S)
__global__ void t_sn_ndp(const float* __restrict__ X, float* __restrict__ Yf,
                         unsigned short* __restrict__ Ybf, int NN, int S, int D) {
  long long i = (long long)blockIdx.x * blockDim.x + threadIdx.x;
  long long tot = (long long)NN * S * D;
  if (i >= tot) return;
  int s = (int)(i % S); long long t = i / S;
  int d = (int)(t % D); int n = (int)(t / D);
  float v = X[((size_t)s * NN + n) * D + d];
  if (Yf)  Yf[i] = v;
  if (Ybf) Ybf[i] = f2bf(v);
}

// sinusoidal positional encoding add, layout (S,B,D)
__global__ void pe_add_k(float* __restrict__ obj, unsigned short* __restrict__ objbf,
                         int S, int B, int D) {
  long long i = (long long)blockIdx.x * blockDim.x + threadIdx.x;
  long long tot = (long long)S * B * D;
  if (i >= tot) return;
  int d = (int)(i % D); long long t = i / D;
  int s = (int)(t / B);
  int i2 = d & ~1;
  float div = __expf(-logf(10000.f) * (float)i2 / (float)D);
  float ang = (float)s * div;
  float pe = (d & 1) ? __cosf(ang) : __sinf(ang);
  float v = obj[i] + pe;
  obj[i] = v;
  objbf[i] = f2bf(v);
}

// LayerNorm over last dim (<=256), one wave32 per row, optional resid/act.
__global__ void ln_k(const float* __restrict__ X, const float* __restrict__ R,
                     const float* __restrict__ g, const float* __restrict__ bta,
                     float* __restrict__ Yf, unsigned short* __restrict__ Ybf,
                     int rows, int Dn, int act) {
  int row = blockIdx.x * 8 + (threadIdx.x >> 5);
  int lane = threadIdx.x & 31;
  if (row >= rows) return;
  const float* x = X + (size_t)row * Dn;
  float v[8];
  int cnt = (Dn + 31) / 32;
  float s = 0.f;
  for (int i = 0; i < cnt; ++i) {
    int j = lane + 32 * i;
    float t = (j < Dn) ? x[j] : 0.f;
    if (R && j < Dn) t += R[(size_t)row * Dn + j];
    v[i] = t; s += t;
  }
  for (int m = 16; m; m >>= 1) s += __shfl_xor(s, m);
  float mean = s / (float)Dn;
  float q = 0.f;
  for (int i = 0; i < cnt; ++i) {
    int j = lane + 32 * i;
    float d = (j < Dn) ? v[i] - mean : 0.f;
    q += d * d;
  }
  for (int m = 16; m; m >>= 1) q += __shfl_xor(q, m);
  float rstd = rsqrtf(q / (float)Dn + 1e-5f);
  for (int i = 0; i < cnt; ++i) {
    int j = lane + 32 * i;
    if (j >= Dn) continue;
    float y = (v[i] - mean) * rstd * g[j] + bta[j];
    y = act_apply(y, act);
    if (Yf)  Yf[(size_t)row * Dn + j] = y;
    if (Ybf) Ybf[(size_t)row * Dn + j] = f2bf(y);
  }
}

// MHA core on fused QKV f32 (token = s*B + b, row width 3D). Writes bf16 out.
__global__ void attn_k(const float* __restrict__ QKV, unsigned short* __restrict__ O,
                       int S, int B, int D, int H) {
  __shared__ float sc[8][132];
  int tid = threadIdx.x, lane = tid & 31, wave = tid >> 5;
  int bh = blockIdx.x, b = bh / H, h = bh % H;
  int dh = D / H;
  float scale = rsqrtf((float)dh);
  int s = blockIdx.y * 8 + wave;
  bool on = (s < S);
  if (on) {
    const float* q = QKV + ((size_t)s * B + b) * 3 * D + h * dh;
    for (int t = 0; t < S; ++t) {
      const float* kk = QKV + ((size_t)t * B + b) * 3 * D + D + h * dh;
      float p = 0.f;
      for (int j = lane; j < dh; j += 32) p += q[j] * kk[j];
      for (int m = 16; m; m >>= 1) p += __shfl_xor(p, m);
      if (lane == 0) sc[wave][t] = p * scale;
    }
  }
  __syncthreads();
  float inv = 0.f;
  if (on) {
    float mx = -3.0e38f;
    for (int t = lane; t < S; t += 32) mx = fmaxf(mx, sc[wave][t]);
    for (int m = 16; m; m >>= 1) mx = fmaxf(mx, __shfl_xor(mx, m));
    float sum = 0.f;
    for (int t = lane; t < S; t += 32) {
      float e = __expf(sc[wave][t] - mx);
      sc[wave][t] = e; sum += e;
    }
    for (int m = 16; m; m >>= 1) sum += __shfl_xor(sum, m);
    inv = 1.f / sum;
  }
  __syncthreads();
  if (on) {
    for (int j = lane; j < dh; j += 32) {
      float o = 0.f;
      for (int t = 0; t < S; ++t)
        o += sc[wave][t] * QKV[((size_t)t * B + b) * 3 * D + 2 * D + h * dh + j];
      O[((size_t)s * B + b) * D + h * dh + j] = f2bf(o * inv);
    }
  }
}

// --------------------------- host orchestration ----------------------------

namespace {
constexpr int NI = 128, D = 256, HW = 28, H2 = 14, H4 = 7, S = 49, FFD = 2048, DD = 64;
constexpr long long EL_ROI  = (long long)NI * D * HW * HW;
constexpr long long EL_ENC1 = (long long)NI * D * H2 * H2;
constexpr long long EL_ENC2 = (long long)NI * D * H4 * H4;
constexpr long long EL_TOK  = (long long)NI * S * D;
constexpr long long EL_QKV  = (long long)NI * S * 3 * D;
constexpr long long EL_HID  = (long long)NI * S * FFD;
constexpr long long EL_PAR  = (long long)NI * 2 * D * DD;
constexpr long long EL_CW   = 9LL * D * D;   // one packed conv-weight layer

inline void pack(hipStream_t st, const float* x, unsigned short* y, long long n) {
  pack_bf16_k<<<dim3((unsigned)((n + 255) / 256)), dim3(256), 0, st>>>(x, y, n);
}
inline void gemm(hipStream_t st, const unsigned short* A, const unsigned short* B,
                 const float* bias, float* Cf, unsigned short* Cbf,
                 int M, int N, int K, int b_kn, int act,
                 long long sA = 0, long long sB = 0, long long sC = 0, int batch = 1) {
  dim3 g((N + 127) / 128, (M + 127) / 128, batch);
  wmma_gemm<<<g, dim3(256), 0, st>>>(A, B, bias, Cf, Cbf, M, N, K, b_kn, act,
                                     sA, sB, sC);
}
inline void conv(hipStream_t st, const unsigned short* X, const unsigned short* Wbf,
                 const float* cb, const float* bng, const float* bnb,
                 const float* resid, float* Yf, unsigned short* Ybf,
                 int Hin, int Ho, int stride, int ups, int act) {
  int P = Ho * Ho;
  dim3 g((P + 127) / 128, D / 128, NI);
  wmma_conv3x3<<<g, dim3(256), 0, st>>>(X, Wbf, cb, bng, bnb, resid, Yf, Ybf,
                                        D, D, Hin, Hin, Ho, Ho, stride, ups, act);
}
inline void ln(hipStream_t st, const float* X, const float* R, const float* g,
               const float* b, float* Yf, unsigned short* Ybf, int rows, int Dn, int act) {
  ln_k<<<dim3((rows + 7) / 8), dim3(256), 0, st>>>(X, R, g, b, Yf, Ybf, rows, Dn, act);
}
}  // namespace

extern "C" void kernel_launch(void* const* d_in, const int* in_sizes, int n_in,
                              void* d_out, int out_size, void* d_ws, size_t ws_size,
                              hipStream_t stream) {
  (void)in_sizes; (void)n_in; (void)out_size; (void)ws_size;
  const float* roi   = (const float*)d_in[0];
  const float* pro0  = (const float*)d_in[1];
  const float* enc_w = (const float*)d_in[3];
  const float* enc_b = (const float*)d_in[4];
  const float* enc_g = (const float*)d_in[5];
  const float* enc_bb = (const float*)d_in[6];
  const float* aqkv_w = (const float*)d_in[7];
  const float* aqkv_b = (const float*)d_in[8];
  const float* aout_w = (const float*)d_in[9];
  const float* aout_b = (const float*)d_in[10];
  const float* n1g = (const float*)d_in[11], *n1b = (const float*)d_in[12];
  const float* n2g = (const float*)d_in[13], *n2b = (const float*)d_in[14];
  const float* n3g = (const float*)d_in[15], *n3b = (const float*)d_in[16];
  const float* dyn_w = (const float*)d_in[17], *dyn_b = (const float*)d_in[18];
  const float* d1g = (const float*)d_in[19], *d1b = (const float*)d_in[20];
  const float* d2g = (const float*)d_in[21], *d2b = (const float*)d_in[22];
  const float* l1w = (const float*)d_in[23], *l1b = (const float*)d_in[24];
  const float* l2w = (const float*)d_in[25], *l2b = (const float*)d_in[26];
  const float* teqw = (const float*)d_in[27], *teqb = (const float*)d_in[28];
  const float* teow = (const float*)d_in[29], *teob = (const float*)d_in[30];
  const float* tl1g = (const float*)d_in[31], *tl1b = (const float*)d_in[32];
  const float* tl2g = (const float*)d_in[33], *tl2b = (const float*)d_in[34];
  const float* tf1w = (const float*)d_in[35], *tf1b = (const float*)d_in[36];
  const float* tf2w = (const float*)d_in[37], *tf2b = (const float*)d_in[38];
  const float* recw = (const float*)d_in[39], *recb = (const float*)d_in[40];
  const float* recg = (const float*)d_in[41], *recbb = (const float*)d_in[42];
  const float* dAw = (const float*)d_in[43], *dAb = (const float*)d_in[44];
  const float* dAg = (const float*)d_in[45], *dAbb = (const float*)d_in[46];
  const float* dBw = (const float*)d_in[47], *dBb = (const float*)d_in[48];
  const float* dFw = (const float*)d_in[49], *dFb = (const float*)d_in[50];
  const float* dFg = (const float*)d_in[51], *dFbb = (const float*)d_in[52];

  char* base = (char*)d_ws;
  size_t off = 0;
  auto A = [&](long long elems, int esz) -> char* {
    char* p = base + off;
    off = (off + (size_t)elems * esz + 255) & ~(size_t)255;
    return p;
  };
  auto F  = [&](long long e) { return (float*)A(e, 4); };
  auto Hb = [&](long long e) { return (unsigned short*)A(e, 2); };

  // bf16 GEMM weights
  unsigned short* wqkv = Hb(768 * 256);
  unsigned short* wout = Hb(256 * 256);
  unsigned short* wdyn = Hb((long long)2 * 16384 * 256);
  unsigned short* wl1  = Hb((long long)FFD * 256);
  unsigned short* wl2  = Hb((long long)256 * FFD);
  unsigned short* wteq = Hb((long long)3 * 768 * 256);
  unsigned short* wteo = Hb((long long)3 * 256 * 256);
  unsigned short* wtf1 = Hb((long long)3 * FFD * 256);
  unsigned short* wtf2 = Hb((long long)3 * 256 * FFD);
  // packed conv weights: enc0,enc1,rec0,rec1,dA0,dA1,dB0,dB1,final
  unsigned short* cwv = Hb(9LL * EL_CW);
  auto CW = [&](int i) { return cwv + (size_t)i * EL_CW; };
  // activations
  unsigned short* roi_bf   = Hb(EL_ROI);
  float*          enc1_f   = F(EL_ENC1);
  unsigned short* enc1_bf  = Hb(EL_ENC1);
  float*          enc2_f   = F(EL_ENC2);
  float*          feats_f  = F(EL_TOK);
  unsigned short* feats_bf = Hb(EL_TOK);
  unsigned short* pro0_bf  = Hb(NI * D);
  float*          pqkv_f   = F(NI * 3 * D);
  unsigned short* proA_bf  = Hb(NI * D);
  float*          proj_f   = F(NI * D);
  float*          pro_f    = F(NI * D);
  unsigned short* pro_bf   = Hb(NI * D);
  float*          par_f    = F(EL_PAR);
  unsigned short* par_bf   = Hb(EL_PAR);
  float*          f1_f     = F((long long)NI * S * DD);
  unsigned short* f1_bf    = Hb((long long)NI * S * DD);
  float*          f2_f     = F(EL_TOK);
  float*          f3_f     = F(EL_TOK);
  float*          obj_f    = F(EL_TOK);
  unsigned short* obj_bf   = Hb(EL_TOK);
  unsigned short* hid_bf   = Hb(EL_HID);
  float*          obj2_f   = F(EL_TOK);
  float*          objA_f   = F(EL_TOK);
  float*          osn_f    = F(EL_TOK);
  unsigned short* osn_bf   = Hb(EL_TOK);
  float*          qkv_f    = F(EL_QKV);
  unsigned short* atO_bf   = Hb(EL_TOK);
  float*          tmp_f    = F(EL_TOK);
  float*          omap_f   = F(EL_ENC2);
  unsigned short* omap_bf  = Hb(EL_ENC2);
  unsigned short* krec0_bf = Hb(EL_ENC2);
  float*          rec1_f   = F(EL_ENC1);
  unsigned short* tA_bf    = Hb(EL_ENC1);
  float*          det1_f   = F(EL_ENC1);
  unsigned short* det1_bf  = Hb(EL_ENC1);
  unsigned short* krec1_bf = Hb(EL_ENC1);
  float*          rec2_f   = F(EL_ROI);
  unsigned short* tA2_bf   = Hb(EL_ROI);
  float*          det2_f   = F(EL_ROI);
  unsigned short* krec2_bf = (unsigned short*)tA2_bf;   // tA2 dead after det2
  float*          fin_f    = F(EL_ROI);

  hipStream_t st = stream;
  auto repack = [&](const float* w, unsigned short* y) {
    repack_convw_k<<<dim3((unsigned)((EL_CW + 255) / 256)), 256, 0, st>>>(w, y, D, D);
  };

  // ---- weight packing ----
  pack(st, aqkv_w, wqkv, 768 * 256);
  pack(st, aout_w, wout, 256 * 256);
  pack(st, dyn_w,  wdyn, (long long)2 * 16384 * 256);
  pack(st, l1w, wl1, (long long)FFD * 256);
  pack(st, l2w, wl2, (long long)256 * FFD);
  pack(st, teqw, wteq, (long long)3 * 768 * 256);
  pack(st, teow, wteo, (long long)3 * 256 * 256);
  pack(st, tf1w, wtf1, (long long)3 * FFD * 256);
  pack(st, tf2w, wtf2, (long long)3 * 256 * FFD);
  repack(enc_w, CW(0));              repack(enc_w + EL_CW, CW(1));
  repack(recw, CW(2));               repack(recw + EL_CW, CW(3));
  repack(dAw, CW(4));                repack(dAw + EL_CW, CW(5));
  repack(dBw, CW(6));                repack(dBw + EL_CW, CW(7));
  repack(dFw, CW(8));

  // ---- encoder pyramid ----
  pack(st, roi, roi_bf, EL_ROI);
  conv(st, roi_bf, CW(0), enc_b, enc_g, enc_bb, nullptr, enc1_f, enc1_bf,
       HW, H2, 2, 0, 1);
  conv(st, enc1_bf, CW(1), enc_b + D, enc_g + D, enc_bb + D, nullptr,
       enc2_f, nullptr, H2, H4, 2, 0, 1);
  t_ndp_npd<<<dim3((unsigned)((EL_TOK + 255) / 256)), 256, 0, st>>>(
      enc2_f, feats_f, feats_bf, NI, D, S);

  // ---- self-attention over query tokens (S=128, B=1, H=8) ----
  pack(st, pro0, pro0_bf, NI * D);
  gemm(st, pro0_bf, wqkv, aqkv_b, pqkv_f, nullptr, NI, 3 * D, D, 0, 0);
  attn_k<<<dim3(8, (NI + 7) / 8), 256, 0, st>>>(pqkv_f, proA_bf, NI, 1, D, 8);
  gemm(st, proA_bf, wout, aout_b, proj_f, nullptr, NI, D, D, 0, 0);
  ln(st, proj_f, pro0, n1g, n1b, pro_f, pro_bf, NI, D, 0);

  // ---- dynamic conv ----
  gemm(st, pro_bf, wdyn, dyn_b, par_f, par_bf, NI, 2 * 16384, D, 0, 0);
  gemm(st, feats_bf, par_bf, nullptr, f1_f, nullptr, S, DD, D, 1, 0,
       (long long)S * D, 2 * 16384, (long long)S * DD, NI);
  ln(st, f1_f, nullptr, d1g, d1b, nullptr, f1_bf, NI * S, DD, 2);
  gemm(st, f1_bf, par_bf + 16384, nullptr, f2_f, nullptr, S, D, DD, 1, 0,
       (long long)S * DD, 2 * 16384, (long long)S * D, NI);
  ln(st, f2_f, nullptr, d2g, d2b, f3_f, nullptr, NI * S, D, 2);
  ln(st, f3_f, feats_f, n2g, n2b, obj_f, obj_bf, NI * S, D, 0);

  // ---- FFN (ELU) ----
  gemm(st, obj_bf, wl1, l1b, nullptr, hid_bf, NI * S, FFD, D, 0, 2);
  gemm(st, hid_bf, wl2, l2b, obj2_f, nullptr, NI * S, D, FFD, 0, 0);
  ln(st, obj2_f, obj_f, n3g, n3b, objA_f, nullptr, NI * S, D, 0);

  // ---- PE + 3-layer transformer encoder (S=49, B=128, H=4) ----
  t_ns_sn<<<dim3((unsigned)((EL_TOK + 255) / 256)), 256, 0, st>>>(objA_f, osn_f, NI, S, D);
  pe_add_k<<<dim3((unsigned)((EL_TOK + 255) / 256)), 256, 0, st>>>(osn_f, osn_bf, S, NI, D);
  for (int l = 0; l < 3; ++l) {
    gemm(st, osn_bf, wteq + (size_t)l * 768 * 256, teqb + l * 768, qkv_f, nullptr,
         NI * S, 3 * D, D, 0, 0);
    attn_k<<<dim3(NI * 4, (S + 7) / 8), 256, 0, st>>>(qkv_f, atO_bf, S, NI, D, 4);
    gemm(st, atO_bf, wteo + (size_t)l * 256 * 256, teob + l * 256, tmp_f, nullptr,
         NI * S, D, D, 0, 0);
    ln(st, tmp_f, osn_f, tl1g + l * 256, tl1b + l * 256, osn_f, osn_bf, NI * S, D, 0);
    gemm(st, osn_bf, wtf1 + (size_t)l * FFD * 256, tf1b + l * FFD, nullptr, hid_bf,
         NI * S, FFD, D, 0, 1);
    gemm(st, hid_bf, wtf2 + (size_t)l * 256 * FFD, tf2b + l * 256, tmp_f, nullptr,
         NI * S, D, FFD, 0, 0);
    ln(st, tmp_f, osn_f, tl2g + l * 256, tl2b + l * 256, osn_f, osn_bf, NI * S, D, 0);
  }
  t_sn_ndp<<<dim3((unsigned)((EL_TOK + 255) / 256)), 256, 0, st>>>(
      osn_f, omap_f, omap_bf, NI, S, D);

  // ---- gated rec/det decoder pyramid ----
  gate_k<<<dim3((unsigned)((EL_ENC2 + 255) / 256)), 256, 0, st>>>(
      enc2_f, omap_f, krec0_bf, EL_ENC2);
  conv(st, krec0_bf, CW(2), recb, recg, recbb, enc1_f, rec1_f, nullptr,
       H4, H2, 1, 1, 1);
  conv(st, omap_bf, CW(4), dAb, dAg, dAbb, nullptr, nullptr, tA_bf,
       H4, H2, 1, 1, 1);
  conv(st, tA_bf, CW(6), dBb, nullptr, nullptr, enc1_f, det1_f, det1_bf,
       H2, H2, 1, 0, 0);
  gate_k<<<dim3((unsigned)((EL_ENC1 + 255) / 256)), 256, 0, st>>>(
      rec1_f, det1_f, krec1_bf, EL_ENC1);
  conv(st, krec1_bf, CW(3), recb + D, recg + D, recbb + D, roi, rec2_f, nullptr,
       H2, HW, 1, 1, 1);
  conv(st, det1_bf, CW(5), dAb + D, dAg + D, dAbb + D, nullptr, nullptr, tA2_bf,
       H2, HW, 1, 1, 1);
  conv(st, tA2_bf, CW(7), dBb + D, nullptr, nullptr, roi, det2_f, nullptr,
       HW, HW, 1, 0, 0);
  gate_k<<<dim3((unsigned)((EL_ROI + 255) / 256)), 256, 0, st>>>(
      rec2_f, det2_f, krec2_bf, EL_ROI);
  conv(st, krec2_bf, CW(8), dFb, dFg, dFbb, nullptr, fin_f, nullptr,
       HW, HW, 1, 0, 1);
  t_ndp_npd<<<dim3((unsigned)((EL_ROI + 255) / 256)), 256, 0, st>>>(
      fin_f, (float*)d_out, nullptr, NI, D, HW * HW);
}
